// QRNNLayer_43009802502834
// MI455X (gfx1250) — matrix operand order repaired and checked by reading, not compiled, not run on hardware
//
#include <hip/hip_runtime.h>
#include <hip/hip_bf16.h>

// ---------------------------------------------------------------------------
// QRNN layer: B=32, T=1024, I=512, H=512, K=2
//   raw[m, n] = [x[b,t-1,:], x[b,t,:]] @ Wcat   (M=32768, Kred=1024, N=1024)
//   f = sigmoid(raw_f), z = sigmoid(raw_z); h[t] = f*h[t-1] + (1-f)*z
// Workspace layout (bytes):
//   [0, 33587200)              Xp   bf16 (B, T+1, I), row t=0 zeroed
//   [33587200, 35684352)       Bcat bf16 (N=1024, K=1024) N-major
//   [35684352, +64MB)          SF   f32  (M, 512)  sigmoid(raw_f)
//   [35684352+64MB, +64MB)     SZ   f32  (M, 512)  sigmoid(raw_z)
// ---------------------------------------------------------------------------

typedef __attribute__((ext_vector_type(16))) __bf16 v16bf;
typedef __attribute__((ext_vector_type(8)))  float  v8f;
typedef __attribute__((ext_vector_type(4)))  int    v4i;

typedef v4i __attribute__((address_space(1)))* g128_t;   // global int4*
typedef v4i __attribute__((address_space(3)))* l128_t;   // LDS int4*

union FragB { uint4 u4[2]; v16bf v; };

#if __has_builtin(__builtin_amdgcn_global_load_async_to_lds_b128)
#define USE_ASYNC_LDS 1
#else
#define USE_ASYNC_LDS 0
#endif

__device__ __forceinline__ void async_b128(const unsigned short* g, unsigned short* l) {
#if USE_ASYNC_LDS
    __builtin_amdgcn_global_load_async_to_lds_b128((g128_t)g, (l128_t)l, 0, 0);
#else
    (void)g; (void)l;
#endif
}

__device__ __forceinline__ void wait_async0() {
#if __has_builtin(__builtin_amdgcn_s_wait_asynccnt)
    __builtin_amdgcn_s_wait_asynccnt(0);
#else
    asm volatile("s_wait_asynccnt 0x0" ::: "memory");
#endif
}

__device__ __forceinline__ unsigned short f2bf(float x) {
    union { float f; unsigned u; } a; a.f = x;
    unsigned u = a.u;
    u += 0x7FFFu + ((u >> 16) & 1u);           // round-to-nearest-even
    return (unsigned short)(u >> 16);
}

// ---------------- pack inputs -> padded bf16 (B, T+1, I) -------------------
__global__ __launch_bounds__(256) void pack_x(const float* __restrict__ X,
                                              unsigned short* __restrict__ Xp) {
    int g = blockIdx.x * blockDim.x + threadIdx.x;     // 2,099,200 threads
    size_t e0 = (size_t)g * 8;
    int rp = (int)(e0 >> 9);                            // b*1025 + tp
    int i0 = (int)(e0 & 511);
    int b  = rp / 1025;
    int tp = rp - b * 1025;
    union { unsigned short s[8]; uint4 u; } t;
    if (tp == 0) {
        t.u = make_uint4(0u, 0u, 0u, 0u);
    } else {
        const float* src = X + ((size_t)(b * 1024 + (tp - 1)) * 512 + i0);
        float4 f0 = *(const float4*)src;
        float4 f1 = *(const float4*)(src + 4);
        t.s[0] = f2bf(f0.x); t.s[1] = f2bf(f0.y); t.s[2] = f2bf(f0.z); t.s[3] = f2bf(f0.w);
        t.s[4] = f2bf(f1.x); t.s[5] = f2bf(f1.y); t.s[6] = f2bf(f1.z); t.s[7] = f2bf(f1.w);
    }
    *(uint4*)&Xp[e0] = t.u;
}

// -------- pack Wf/Wz (H,I,2) -> Bcat (N=1024 rows, K=1024) bf16 ------------
__global__ __launch_bounds__(256) void pack_w(const float* __restrict__ Wf,
                                              const float* __restrict__ Wz,
                                              unsigned short* __restrict__ Bt) {
    int g  = blockIdx.x * blockDim.x + threadIdx.x;    // 131072 threads
    int k0 = (g & 127) * 8;
    int n  = g >> 7;
    const float* W = (n < 512) ? Wf : Wz;
    int h   = n & 511;
    int sel = k0 >> 9;                                  // which K tap
    int kk  = k0 & 511;
    union { unsigned short s[8]; uint4 u; } t;
    #pragma unroll
    for (int j = 0; j < 8; ++j)
        t.s[j] = f2bf(W[(size_t)h * 1024 + (size_t)(kk + j) * 2 + sel]);
    *(uint4*)&Bt[(size_t)n * 1024 + k0] = t.u;
}

// ------------------------- WMMA bf16 GEMM ----------------------------------
#define TILE_M 128
#define TILE_N 128
#define TILE_K 32
#define LDT    40   // TILE_K + 8 pad (halfs)

__global__ __launch_bounds__(256) void qrnn_gemm(
    const unsigned short* __restrict__ Xp,   // bf16 bits, (B,1025,512)
    const unsigned short* __restrict__ Bt,   // bf16 bits, (1024,1024) N-major
    const float* __restrict__ biasF, const float* __restrict__ biasZ,
    float* __restrict__ SF, float* __restrict__ SZ)
{
    __shared__ __attribute__((aligned(16))) unsigned short Ash[2][TILE_M][LDT];
    __shared__ __attribute__((aligned(16))) unsigned short Bsh[2][TILE_N][LDT];

    const int tid  = threadIdx.x;
    const int lane = tid & 31;
    const int wave = tid >> 5;
    const int wm   = wave & 1;          // 2 waves along M
    const int wn   = wave >> 1;         // 4 waves along N

    const int n0 = blockIdx.x * TILE_N;
    const int m0 = blockIdx.y * TILE_M;
    const int b  = m0 >> 10;
    const int t0 = m0 & 1023;

    // cooperative tile loads: thread -> (row, 16-half chunk)
    const int lrow = tid >> 1;
    const int lofs = (tid & 1) * 16;
    const unsigned short* agp = Xp + ((size_t)(b * 1025 + t0 + lrow)) * 512 + lofs;
    const unsigned short* bgp = Bt + ((size_t)(n0 + lrow)) * 1024 + lofs;

    v8f acc[4][2] = {};

    // preload K-tile 0 straight into LDS
#if USE_ASYNC_LDS
    async_b128(agp,     &Ash[0][lrow][lofs]);
    async_b128(agp + 8, &Ash[0][lrow][lofs + 8]);
    async_b128(bgp,     &Bsh[0][lrow][lofs]);
    async_b128(bgp + 8, &Bsh[0][lrow][lofs + 8]);
    wait_async0();
#else
    {
        uint4 a0 = *(const uint4*)(agp);
        uint4 a1 = *(const uint4*)(agp + 8);
        uint4 b0 = *(const uint4*)(bgp);
        uint4 b1 = *(const uint4*)(bgp + 8);
        *(uint4*)&Ash[0][lrow][lofs]     = a0;
        *(uint4*)&Ash[0][lrow][lofs + 8] = a1;
        *(uint4*)&Bsh[0][lrow][lofs]     = b0;
        *(uint4*)&Bsh[0][lrow][lofs + 8] = b1;
    }
#endif
    __syncthreads();

    const int lr16 = lane & 15;
    const int sel  = lane >> 4;         // 0 / 1
    const int acb  = sel * 8;           // A per-lane chunk base (halfs)
    const int bkb  = sel * 8;           // B per-lane chunk base (halfs)

    int buf = 0;
    const int NKT = 1024 / TILE_K;      // 32
    for (int kt = 0; kt < NKT; ++kt) {
        const bool more = (kt + 1) < NKT;
#if USE_ASYNC_LDS
        if (more) {
            const int k0 = (kt + 1) * TILE_K;
            async_b128(agp + k0,     &Ash[buf ^ 1][lrow][lofs]);
            async_b128(agp + k0 + 8, &Ash[buf ^ 1][lrow][lofs + 8]);
            async_b128(bgp + k0,     &Bsh[buf ^ 1][lrow][lofs]);
            async_b128(bgp + k0 + 8, &Bsh[buf ^ 1][lrow][lofs + 8]);
        }
#else
        uint4 ra0, ra1, rb0, rb1;
        if (more) {
            const int k0 = (kt + 1) * TILE_K;
            ra0 = *(const uint4*)(agp + k0);
            ra1 = *(const uint4*)(agp + k0 + 8);
            rb0 = *(const uint4*)(bgp + k0);
            rb1 = *(const uint4*)(bgp + k0 + 8);
        }
#endif

        FragB afr[4], bfr[2];
        #pragma unroll
        for (int i = 0; i < 4; ++i) {
            const unsigned short* p = &Ash[buf][wm * 64 + i * 16 + lr16][acb];
            afr[i].u4[0] = *(const uint4*)p;          // K acb..acb+7
            afr[i].u4[1] = *(const uint4*)(p + 16);   // K acb+16..acb+23
        }
        #pragma unroll
        for (int j = 0; j < 2; ++j) {
            const unsigned short* p = &Bsh[buf][wn * 32 + j * 16 + lr16][bkb];
            bfr[j].u4[0] = *(const uint4*)p;          // K bkb..bkb+7
            bfr[j].u4[1] = *(const uint4*)(p + 16);   // K bkb+16..bkb+23
        }

        #pragma unroll
        for (int i = 0; i < 4; ++i)
            #pragma unroll
            for (int j = 0; j < 2; ++j)
                acc[i][j] = __builtin_amdgcn_wmma_f32_16x16x32_bf16(
                    false, afr[i].v, false, bfr[j].v,
                    (short)0, acc[i][j], false, false);

#if USE_ASYNC_LDS
        if (more) wait_async0();
#else
        if (more) {
            *(uint4*)&Ash[buf ^ 1][lrow][lofs]     = ra0;
            *(uint4*)&Ash[buf ^ 1][lrow][lofs + 8] = ra1;
            *(uint4*)&Bsh[buf ^ 1][lrow][lofs]     = rb0;
            *(uint4*)&Bsh[buf ^ 1][lrow][lofs + 8] = rb1;
        }
#endif
        __syncthreads();
        buf ^= 1;
    }

    // epilogue: bias + sigmoid, write gate plane
    const int gate = (n0 >= 512);
    const float* bias = gate ? biasZ : biasF;
    float* outp = gate ? SZ : SF;
    #pragma unroll
    for (int i = 0; i < 4; ++i) {
        const int mrow0 = m0 + wm * 64 + i * 16 + sel * 8;
        #pragma unroll
        for (int j = 0; j < 2; ++j) {
            const int ncol = n0 + wn * 32 + j * 16 + lr16;
            const int h = ncol & 511;
            const float bv = bias[h];
            #pragma unroll
            for (int r = 0; r < 8; ++r) {
                float x = acc[i][j][r] + bv;
                float s = 1.0f / (1.0f + __expf(-x));
                outp[(size_t)(mrow0 + r) * 512 + h] = s;
            }
        }
    }
}

// ---------------- sequential gate scan: h[t]=f*h[t-1]+(1-f)*z --------------
__global__ __launch_bounds__(256) void scan_k(const float* __restrict__ SF,
                                              const float* __restrict__ SZ,
                                              float* __restrict__ out) {
    int idx = blockIdx.x * blockDim.x + threadIdx.x;   // 16384 = B*H
    int h = idx & 511;                                  // adjacent lanes -> adjacent h
    int b = idx >> 9;
    const size_t base = (size_t)b * 1024 * 512 + h;
    float hv = 0.0f;
    #pragma unroll 8
    for (int t = 0; t < 1024; ++t) {
        size_t o = base + (size_t)t * 512;
        if ((t & 7) == 0 && t + 32 < 1024) {
            __builtin_prefetch(&SF[o + 32 * 512], 0, 1);   // global_prefetch_b8
            __builtin_prefetch(&SZ[o + 32 * 512], 0, 1);
        }
        float f = SF[o];
        float z = SZ[o];
        hv = fmaf(f, hv, (1.0f - f) * z);
        out[o] = hv;
    }
}

// ---------------------------------------------------------------------------
extern "C" void kernel_launch(void* const* d_in, const int* in_sizes, int n_in,
                              void* d_out, int out_size, void* d_ws, size_t ws_size,
                              hipStream_t stream) {
    const float* x   = (const float*)d_in[0];
    // d_in[1] = init_state (zeros; unused by reference)
    const float* Wz  = (const float*)d_in[2];
    const float* bz  = (const float*)d_in[3];
    const float* Wf  = (const float*)d_in[4];
    const float* bfp = (const float*)d_in[5];

    char* ws = (char*)d_ws;
    unsigned short* Xp = (unsigned short*)(ws);                       // 33,587,200 B
    unsigned short* Bt = (unsigned short*)(ws + 33587200);            //  2,097,152 B
    float* SF = (float*)(ws + 35684352);                              // 67,108,864 B
    float* SZ = (float*)(ws + 35684352 + 67108864);                   // 67,108,864 B
    float* out = (float*)d_out;

    pack_x<<<8200, 256, 0, stream>>>(x, Xp);
    pack_w<<<512, 256, 0, stream>>>(Wf, Wz, Bt);
    qrnn_gemm<<<dim3(8, 256), 256, 0, stream>>>(Xp, Bt, bfp, bz, SF, SZ);
    scan_k<<<64, 256, 0, stream>>>(SF, SZ, out);
}
